// AttentionBlock_34823594836196
// MI455X (gfx1250) — matrix operand order repaired and checked by reading, not compiled, not run on hardware
//
#include <hip/hip_runtime.h>
#include <hip/hip_bf16.h>
#include <stdint.h>

typedef __attribute__((ext_vector_type(16))) __bf16 v16bf;
typedef __attribute__((ext_vector_type(8)))  float  v8f;

// ---------------------------------------------------------------------------
// WMMA GEMM: C[M,N] = A[M,K] x B + optional bias (bias over N).
// TRANS_B=true : B stored as [N,K]  (i.e. computes A @ Bmat^T, torch Linear)
// TRANS_B=false: B stored as [K,N]
// One wave per 16x16 C tile, K-step 32, bf16 inputs / f32 accumulate.
// ---------------------------------------------------------------------------
template <bool TRANS_B, bool HAS_BIAS>
__global__ __launch_bounds__(256) void wmma_gemm_kernel(
    const float* __restrict__ A, const float* __restrict__ B,
    const float* __restrict__ bias, float* __restrict__ C,
    int M, int N, int K) {
  const int wave = threadIdx.x >> 5;
  const int lane = threadIdx.x & 31;
  const int half = lane >> 4;   // 0: lanes 0-15, 1: lanes 16-31
  const int row  = lane & 15;

  const int tiles_n = N >> 4;
  const int tile = blockIdx.x * 8 + wave;      // 8 waves per block
  const int tm = (tile / tiles_n) << 4;
  const int tn = (tile % tiles_n) << 4;
  if (tm >= M) return;                          // wave-uniform exit

  v8f acc = {};
  for (int k0 = 0; k0 < K; k0 += 32) {
    v16bf a, b;
    // A fragment: this lane holds M = tm+row.
    // elements 0..7  -> K = k0 + half*16 + e       (contiguous 8 floats)
    // elements 8..15 -> K = k0 + half*16 + 8 + e   (contiguous 8 floats)
    const float* ap = A + (size_t)(tm + row) * K + k0 + half * 16;
#pragma unroll
    for (int e = 0; e < 8; ++e) a[e]     = (__bf16)ap[e];
#pragma unroll
    for (int e = 0; e < 8; ++e) a[8 + e] = (__bf16)ap[8 + e];

    // B fragment: this lane holds N-col = tn+row; element e -> K = k0+half*16+e
    if (TRANS_B) {
      const float* bp = B + (size_t)(tn + row) * K + k0 + half * 16;
#pragma unroll
      for (int e = 0; e < 16; ++e) b[e] = (__bf16)bp[e];
    } else {
      const float* bp = B + (size_t)(k0 + half * 16) * N + (tn + row);
#pragma unroll
      for (int e = 0; e < 16; ++e) b[e] = (__bf16)bp[(size_t)e * N];
    }
    acc = __builtin_amdgcn_wmma_f32_16x16x32_bf16(
        /*neg_a=*/false, a, /*neg_b=*/false, b,
        /*c_mod=*/(short)0, acc, /*reuse_a=*/false, /*reuse_b=*/false);
  }

  const float bv = HAS_BIAS ? bias[tn + row] : 0.0f;
  // D layout: VGPR i -> M = tm + i + half*8, N = tn + row
#pragma unroll
  for (int i = 0; i < 8; ++i)
    C[(size_t)(tm + i + half * 8) * N + (tn + row)] = acc[i] + bv;
}

// ---------------------------------------------------------------------------
// Streaming single-query attention over X = past_predictions, one HBM pass,
// double-buffered via GLOBAL_LOAD_ASYNC_TO_LDS_B128 (ASYNCcnt-tracked).
// Per block b: scores s_t = r[b] . X[b,t,:]; online softmax; u[b] = w . X[b].
// ---------------------------------------------------------------------------
#define FDIM 1024
#define TCH  16     // rows of X per LDS tile (64 KB); 2 buffers + r = 132 KB

__device__ __forceinline__ void async_copy_b128(uint32_t lds_addr,
                                                const float* gaddr) {
  asm volatile("global_load_async_to_lds_b128 %0, %1, off"
               :: "v"(lds_addr), "v"(gaddr) : "memory");
}

__global__ __launch_bounds__(256) void attn_stream_kernel(
    const float* __restrict__ X,   // [B, T, F]
    const float* __restrict__ R,   // [B, F]
    float* __restrict__ U,         // [B, F]
    int T) {
  __shared__ float r_s[FDIM];
  __shared__ float x_s[2][TCH * FDIM];
  __shared__ float s_s[TCH];

  const int b    = blockIdx.x;
  const int tid  = threadIdx.x;
  const int lane = tid & 31;
  const int wave = tid >> 5;
  const float* Xb = X + (size_t)b * T * FDIM;

  // stage r[b] into LDS (ordinary loads; 4 KB, one-time)
  for (int i = tid; i < FDIM / 4; i += 256)
    ((float4*)r_s)[i] = ((const float4*)(R + (size_t)b * FDIM))[i];

  // issue async tile c -> x_s[buf]; exactly 16 b128 copies per thread
  auto issue_tile = [&](int c, int buf) {
    const float* src = Xb + (size_t)c * TCH * FDIM;
#pragma unroll
    for (int i = 0; i < 16; ++i) {
      const int e = tid + i * 256;               // float4 index within tile
      uint32_t lds = (uint32_t)(size_t)(void*)&x_s[buf][e * 4];
      async_copy_b128(lds, src + (size_t)e * 4);
    }
  };

  const int nch = T / TCH;                       // 16 chunks
  issue_tile(0, 0);

  float m = -1e30f, l = 0.0f;
  float4 acc = {0.f, 0.f, 0.f, 0.f};
  const int g = tid * 4;                         // 4 output columns per thread

  for (int c = 0; c < nch; ++c) {
    const int cur = c & 1;
    if (c + 1 < nch) {
      issue_tile(c + 1, cur ^ 1);
      // 16 in flight for next tile; <=16 outstanding means tile c has landed
      asm volatile("s_wait_asynccnt 16" ::: "memory");
    } else {
      asm volatile("s_wait_asynccnt 0" ::: "memory");
    }
    __syncthreads();                             // all waves' tile-c data in LDS

    // scores: wave w computes rows w*2 .. w*2+1 via lane-strided dot + shuffle
#pragma unroll
    for (int j = 0; j < 2; ++j) {
      const int t = wave * 2 + j;
      const float* xr = &x_s[cur][t * FDIM];
      float p = 0.f;
#pragma unroll 8
      for (int i = lane; i < FDIM; i += 32) p = fmaf(xr[i], r_s[i], p);
#pragma unroll
      for (int off = 16; off > 0; off >>= 1) p += __shfl_xor(p, off, 32);
      if (lane == 0) s_s[t] = p;
    }
    __syncthreads();

    // online softmax update (replicated per thread; all threads identical)
    float cmax = -1e30f;
#pragma unroll
    for (int t = 0; t < TCH; ++t) cmax = fmaxf(cmax, s_s[t]);
    const float newm  = fmaxf(m, cmax);
    const float scale = __expf(m - newm);
    l *= scale;
    acc.x *= scale; acc.y *= scale; acc.z *= scale; acc.w *= scale;
#pragma unroll
    for (int t = 0; t < TCH; ++t) {
      const float w = __expf(s_s[t] - newm);
      l += w;
      const float4 xv = *(const float4*)(&x_s[cur][t * FDIM + g]);
      acc.x = fmaf(w, xv.x, acc.x);
      acc.y = fmaf(w, xv.y, acc.y);
      acc.z = fmaf(w, xv.z, acc.z);
      acc.w = fmaf(w, xv.w, acc.w);
    }
    m = newm;
    __syncthreads();   // all reads of x_s[cur]/s_s done before reuse
  }

  const float inv = 1.0f / l;
  float4 out = {acc.x * inv, acc.y * inv, acc.z * inv, acc.w * inv};
  *(float4*)(U + (size_t)b * FDIM + g) = out;
}

// ---------------------------------------------------------------------------
extern "C" void kernel_launch(void* const* d_in, const int* in_sizes, int n_in,
                              void* d_out, int out_size, void* d_ws, size_t ws_size,
                              hipStream_t stream) {
  const int B = 512, T = 256, F = 1024;
  const float* z  = (const float*)d_in[0];  // [B,F]
  const float* X  = (const float*)d_in[1];  // [B,T,F]
  const float* Wq = (const float*)d_in[2];  // [F,F]
  const float* Wk = (const float*)d_in[3];  // [F,F]
  const float* Wv = (const float*)d_in[4];  // [F,F]
  const float* bq = (const float*)d_in[5];  // [F]
  // bk (d_in[6]) drops out exactly: softmax is shift-invariant per batch row.
  const float* bv = (const float*)d_in[7];  // [F]
  float* out = (float*)d_out;               // [B,F]

  float* Q = (float*)d_ws;                          // [B,F] (2 MB)
  float* R = (float*)((char*)d_ws + (size_t)B * F * sizeof(float)); // [B,F]
  float* U = Q;                                     // reuse Q slot after K2

  const int tiles  = (B / 16) * (F / 16);   // 2048
  const int blocks = tiles / 8;             // 8 waves (16x16 tiles) per block

  // K1: Q = z @ Wq^T + bq
  wmma_gemm_kernel<true, true><<<blocks, 256, 0, stream>>>(z, Wq, bq, Q, B, F, F);
  // K2: R = Q @ Wk
  wmma_gemm_kernel<false, false><<<blocks, 256, 0, stream>>>(Q, Wk, nullptr, R, B, F, F);
  // K3: streaming online-softmax over X -> U (single HBM pass over 512 MB)
  attn_stream_kernel<<<B, 256, 0, stream>>>(X, R, U, T);
  // K4: out = U @ Wv^T + bv
  wmma_gemm_kernel<true, true><<<blocks, 256, 0, stream>>>(U, Wv, bv, out, B, F, F);
}
